// Model_53429393162914
// MI455X (gfx1250) — matrix operand order repaired and checked by reading, not compiled, not run on hardware
//
#include <hip/hip_runtime.h>
#include <math.h>

typedef __attribute__((ext_vector_type(16))) _Float16 v16h;
typedef __attribute__((ext_vector_type(8)))  _Float16 v8h;
typedef __attribute__((ext_vector_type(8)))  float    v8f;

#define DMOD   512
#define NDEPTH 6
#define NH     8
#define DHD    64
#define FFD    2048
#define MDIM   266
#define MPAD   272
#define NB     4
#define LSEQ   2048
#define ROWS   (NB*LSEQ)
#define DSD    64
#define OSD    64
typedef long long ll;

// ---------------------------------------------------------------------------
// Generalized strided batched GEMM: C = act(alpha*A*B [*rowScale] [+bias]) [+residual]
// fp32 in/out, f16 WMMA compute. 128 threads (4 wave32), tile 64x64, BK=32,
// double-buffered LDS, one barrier per K-step, branch-free clamped staging.
// AK1/BK1: compile-time "K is contiguous" fast paths (float4 global loads).
// ---------------------------------------------------------------------------
struct GemmP {
  const float* A; const float* B; float* C;
  const float* bias; const float* rowScale; const float* residual;
  int M, N, K;
  int ldaM, ldaK, ldbK, ldbN, ldcM, ldcN;
  int batchInner, act;                 // act: 0 none, 1 gelu, 2 sin+eps, 3 atan
  ll  sAo, sAi, sBo, sBi, sCo, sCi, sRsO, sRsI;
  float alpha;
};

#define F4TO(dst, q, o) { dst[(o)]=q.x; dst[(o)+1]=q.y; dst[(o)+2]=q.z; dst[(o)+3]=q.w; }

template<int AK1, int BK1>
__global__ __launch_bounds__(128) void gemm_f16_wmma(GemmP p) {
  __shared__ __align__(16) _Float16 As[2][64][40];   // [buf][m][k], +8 halfs pad
  __shared__ __align__(16) _Float16 Bs[2][64][40];   // [buf][n][k] (K-contig per column)
  const int tid  = threadIdx.x;
  const int wave = tid >> 5, lane = tid & 31;
  const int lh = lane >> 4, l16 = lane & 15;
  const int bb = blockIdx.z;
  const int bo = bb / p.batchInner, bi = bb % p.batchInner;
  const float* A = p.A + bo * p.sAo + bi * p.sAi;
  const float* B = p.B + bo * p.sBo + bi * p.sBi;
  float*       C = p.C + bo * p.sCo + bi * p.sCi;
  const int tileM = blockIdx.y * 64, tileN = blockIdx.x * 64;
  const int wm = (wave >> 1) * 32, wn = (wave & 1) * 32;
  // staging: thread t stages A row sr and B column sr for k-half sh (16 k each);
  // consecutive lanes -> consecutive rows/cols -> coalesced global access.
  const int sr = tid & 63, sh = tid >> 6;

  float va[16], vb[16];

  auto load_stage = [&](int k0) {
    const int kb = k0 + sh * 16;
    { // ---- A row sr, k = kb..kb+15
      int gm = tileM + sr;
      bool rok = gm < p.M;
      if (AK1 && rok && (kb + 16 <= p.K)) {
        const float4* s = (const float4*)(A + (ll)gm * p.ldaM + kb);
        float4 q0 = s[0], q1 = s[1], q2 = s[2], q3 = s[3];
        F4TO(va, q0, 0) F4TO(va, q1, 4) F4TO(va, q2, 8) F4TO(va, q3, 12)
      } else {
#pragma unroll
        for (int e = 0; e < 16; e++) {
          int gk = kb + e;
          bool ok = rok && (gk < p.K);
          ll idx = (ll)(rok ? gm : 0) * p.ldaM + (ll)(ok ? gk : 0) * p.ldaK;
          float v = A[idx];                 // always-valid clamped address
          va[e] = ok ? v : 0.0f;            // select, no branch
        }
      }
    }
    { // ---- B column sr, k = kb..kb+15
      int gn = tileN + sr;
      bool cok = gn < p.N;
      if (BK1 && cok && (kb + 16 <= p.K)) {
        const float4* s = (const float4*)(B + (ll)gn * p.ldbN + kb);
        float4 q0 = s[0], q1 = s[1], q2 = s[2], q3 = s[3];
        F4TO(vb, q0, 0) F4TO(vb, q1, 4) F4TO(vb, q2, 8) F4TO(vb, q3, 12)
      } else {
#pragma unroll
        for (int e = 0; e < 16; e++) {
          int gk = kb + e;
          bool ok = cok && (gk < p.K);
          ll idx = (ll)(ok ? gk : 0) * p.ldbK + (ll)(cok ? gn : 0) * p.ldbN;
          float v = B[idx];
          vb[e] = ok ? v : 0.0f;
        }
      }
    }
    // speculative prefetch two K-tiles ahead (-> global_prefetch_b8)
    if (k0 + 64 < p.K) {
      int gm = tileM + sr; gm = gm < p.M ? gm : 0;
      int gn = tileN + sr; gn = gn < p.N ? gn : 0;
      __builtin_prefetch(A + (ll)gm * p.ldaM + (ll)(kb + 64) * p.ldaK, 0, 1);
      __builtin_prefetch(B + (ll)(kb + 64) * p.ldbK + (ll)gn * p.ldbN, 0, 1);
    }
  };

  auto store_stage = [&](int buf) {
    v8h h0, h1, g0, g1;
#pragma unroll
    for (int e = 0; e < 8; e++) {
      h0[e] = (_Float16)va[e];  h1[e] = (_Float16)va[8 + e];
      g0[e] = (_Float16)vb[e];  g1[e] = (_Float16)vb[8 + e];
    }
    *(v8h*)&As[buf][sr][sh * 16]     = h0;
    *(v8h*)&As[buf][sr][sh * 16 + 8] = h1;
    *(v8h*)&Bs[buf][sr][sh * 16]     = g0;
    *(v8h*)&Bs[buf][sr][sh * 16 + 8] = g1;
  };

  v8f acc[2][2] = {};
  load_stage(0);
  int buf = 0;
  for (int k0 = 0; k0 < p.K; k0 += 32) {
    store_stage(buf);
    __syncthreads();                       // orders store(buf) vs compute(buf) AND
                                           // compute(buf^1)@i-1 vs store(buf^1)@i+1
    if (k0 + 32 < p.K) load_stage(k0 + 32);  // overlap next tile's loads with WMMA

    // Fragments per ISA 16-bit layouts (wave32):
    // A 16x32: lanes0-15 row M=l16 K{0..7,16..23}; lanes16-31 K{8..15,24..31}
    // B 32x16: lanes0-15 col N=l16 K{0..7,16..23}; lanes16-31 K{8..15,24..31}
    v16h af[2], bf[2];
#pragma unroll
    for (int i = 0; i < 2; i++) {
      int row = wm + i * 16 + l16;
      v8h alo = *(const v8h*)&As[buf][row][lh * 8];
      v8h ahi = *(const v8h*)&As[buf][row][16 + lh * 8];
#pragma unroll
      for (int e = 0; e < 8; e++) { af[i][e] = alo[e]; af[i][8 + e] = ahi[e]; }
      int col = wn + i * 16 + l16;
      v8h blo = *(const v8h*)&Bs[buf][col][lh * 8];
      v8h bhi = *(const v8h*)&Bs[buf][col][16 + lh * 8];
#pragma unroll
      for (int e = 0; e < 8; e++) { bf[i][e] = blo[e]; bf[i][8 + e] = bhi[e]; }
    }
#pragma unroll
    for (int i = 0; i < 2; i++)
#pragma unroll
      for (int j = 0; j < 2; j++)
        acc[i][j] = __builtin_amdgcn_wmma_f32_16x16x32_f16(
            false, af[i], false, bf[j], (short)0, acc[i][j], false, false);
    buf ^= 1;
  }

  const float* rs  = p.rowScale ? p.rowScale + bo * p.sRsO + bi * p.sRsI : nullptr;
  const float* res = p.residual ? p.residual + bo * p.sCo + bi * p.sCi : nullptr;
#pragma unroll
  for (int i = 0; i < 2; i++)
#pragma unroll
    for (int j = 0; j < 2; j++)
#pragma unroll
      for (int r = 0; r < 8; r++) {
        int row = tileM + wm + i * 16 + lh * 8 + r;   // C: VGPR r, lane-half selects M+8
        int col = tileN + wn + j * 16 + l16;
        if (row < p.M && col < p.N) {
          float v = acc[i][j][r] * p.alpha;
          if (rs)      v *= rs[row];
          if (p.bias)  v += p.bias[col];
          if      (p.act == 1) v = 0.5f * v * (1.0f + erff(v * 0.70710678118f));
          else if (p.act == 2) v = sinf(v) + 1e-3f;
          else if (p.act == 3) v = atanf(v) * 0.31830988618f + 0.5f;
          ll ci = (ll)row * p.ldcM + (ll)col * p.ldcN;
          if (res) v += res[ci];
          C[ci] = v;
        }
      }
}

// ---------------------------------------------------------------------------
// Elementwise / reduction helper kernels
// ---------------------------------------------------------------------------
__global__ void conv_embed_kernel(const float* x1, const float* w, const float* b, float* x) {
  ll idx = (ll)blockIdx.x * blockDim.x + threadIdx.x;
  ll total = (ll)NB * LSEQ * DMOD;
  if (idx >= total) return;
  int c = (int)(idx % DMOD);
  ll  t = idx / DMOD;
  int l = (int)(t % LSEQ);
  int bb = (int)(t / LSEQ);
  const float* xi = x1 + ((ll)bb * (LSEQ + 1) + l) * 2;
  const float* wc = w + c * 4;
  x[idx] = wc[0] * xi[0] + wc[1] * xi[1] + wc[2] * xi[2] + wc[3] * xi[3] + b[c];
}

__global__ __launch_bounds__(256) void scalenorm_kernel(const float* x, float* o, const float* g, int D) {
  __shared__ float red[8];
  ll row = blockIdx.x;
  const float* xr = x + row * D;
  float s = 0.0f;
  for (int i = threadIdx.x; i < D; i += 256) { float v = xr[i]; s += v * v; }
  for (int off = 16; off > 0; off >>= 1) s += __shfl_down(s, off, 32);
  if ((threadIdx.x & 31) == 0) red[threadIdx.x >> 5] = s;
  __syncthreads();
  if (threadIdx.x == 0) { float t = 0; for (int i = 0; i < 8; i++) t += red[i]; red[0] = t; }
  __syncthreads();
  float n = sqrtf(red[0]) * (1.0f / sqrtf((float)D));
  float scale = g[0] / fmaxf(n, 1e-5f);
  for (int i = threadIdx.x; i < D; i += 256) o[row * D + i] = xr[i] * scale;
}

__global__ void ksum_kernel(const float* kp, float* ks) {
  int m = blockIdx.x * blockDim.x + threadIdx.x;
  int bb = blockIdx.y;
  if (m >= MDIM) return;
  const float* base = kp + (ll)bb * LSEQ * MPAD + m;
  float s = 0.0f;
  for (int i = 0; i < LSEQ; i++) s += base[(ll)i * MPAD];
  ks[(ll)bb * MPAD + m] = s;
}

__global__ void dinv_kernel(const float* qp, const float* ks, float* dinv) {
  ll idx = (ll)blockIdx.x * blockDim.x + threadIdx.x;
  if (idx >= (ll)NB * NH * LSEQ) return;
  ll bb = idx / LSEQ;
  const float* qrow = qp + idx * MPAD;
  const float* k = ks + bb * MPAD;
  float s = 0.0f;
  for (int m = 0; m < MDIM; m++) s += qrow[m] * k[m];
  dinv[idx] = 1.0f / s;
}

// ---------------------------------------------------------------------------
// Host-side orchestration
// ---------------------------------------------------------------------------
static void launch_gemm(hipStream_t st, const float* A, const float* B, float* C,
    int M, int N, int K, int ldaM, int ldaK, int ldbK, int ldbN, int ldcM, int ldcN,
    int batchOuter, int batchInner,
    ll sAo, ll sAi, ll sBo, ll sBi, ll sCo, ll sCi,
    const float* bias, const float* rowScale, ll sRsO, ll sRsI,
    const float* residual, int act, float alpha) {
  GemmP p;
  p.A = A; p.B = B; p.C = C; p.bias = bias; p.rowScale = rowScale; p.residual = residual;
  p.M = M; p.N = N; p.K = K;
  p.ldaM = ldaM; p.ldaK = ldaK; p.ldbK = ldbK; p.ldbN = ldbN; p.ldcM = ldcM; p.ldcN = ldcN;
  p.batchInner = batchInner; p.act = act;
  p.sAo = sAo; p.sAi = sAi; p.sBo = sBo; p.sBi = sBi; p.sCo = sCo; p.sCi = sCi;
  p.sRsO = sRsO; p.sRsI = sRsI; p.alpha = alpha;
  dim3 grid((N + 63) / 64, (M + 63) / 64, batchOuter * batchInner);
  bool ak1 = (ldaK == 1), bk1 = (ldbK == 1);
  if (ak1 && bk1)      gemm_f16_wmma<1, 1><<<grid, 128, 0, st>>>(p);
  else if (ak1)        gemm_f16_wmma<1, 0><<<grid, 128, 0, st>>>(p);
  else                 gemm_f16_wmma<0, 0><<<grid, 128, 0, st>>>(p);
}

struct WS { float *x, *y, *xn, *q, *k, *v, *attno, *qp, *kp, *ctx, *ksum, *dinv, *ffb; };

static void attention(hipStream_t st, WS& w, const float* xq, const float* xkv,
                      const float* qkvw, const float* qkvb,
                      const float* outw, const float* outb,
                      const float* proj, float* io) {
  // q/k/v projections  [8192,512] x [512,512]
  launch_gemm(st, xq,  qkvw,                 w.q, ROWS, DMOD, DMOD, DMOD,1, DMOD,1, DMOD,1,
              1,1, 0,0,0,0,0,0, qkvb,          nullptr,0,0, nullptr, 0, 1.0f);
  launch_gemm(st, xkv, qkvw + DMOD*DMOD,     w.k, ROWS, DMOD, DMOD, DMOD,1, DMOD,1, DMOD,1,
              1,1, 0,0,0,0,0,0, qkvb + DMOD,   nullptr,0,0, nullptr, 0, 1.0f);
  launch_gemm(st, xkv, qkvw + 2*DMOD*DMOD,   w.v, ROWS, DMOD, DMOD, DMOD,1, DMOD,1, DMOD,1,
              1,1, 0,0,0,0,0,0, qkvb + 2*DMOD, nullptr,0,0, nullptr, 0, 1.0f);
  // phi(q), phi(k): per (b,h): [L,64] x projT[64,266] -> sin(alpha*z)+eps
  const float ph_alpha = 0.35355339059f;  // 64^-0.25
  launch_gemm(st, w.q, proj, w.qp, LSEQ, MDIM, DHD, DMOD,1, 1,DHD, MPAD,1,
              NB, NH, (ll)LSEQ*DMOD, DHD, 0,0, (ll)NH*LSEQ*MPAD, (ll)LSEQ*MPAD,
              nullptr, nullptr,0,0, nullptr, 2, ph_alpha);
  launch_gemm(st, w.k, proj, w.kp, LSEQ, MDIM, DHD, DMOD,1, 1,DHD, MPAD,1,
              NB, NH, (ll)LSEQ*DMOD, DHD, 0,0, (ll)NH*LSEQ*MPAD, (ll)LSEQ*MPAD,
              nullptr, nullptr,0,0, nullptr, 2, ph_alpha);
  // k_sum over tokens
  ksum_kernel<<<dim3((MDIM + 127) / 128, NB * NH), 128, 0, st>>>(w.kp, w.ksum);
  // ctx = kp^T @ v  per (b,h): [266, L] x [L, 64]
  launch_gemm(st, w.kp, w.v, w.ctx, MDIM, DHD, LSEQ, 1,MPAD, DMOD,1, DHD,1,
              NB, NH, (ll)NH*LSEQ*MPAD, (ll)LSEQ*MPAD, (ll)LSEQ*DMOD, DHD,
              (ll)NH*MPAD*DHD, (ll)MPAD*DHD,
              nullptr, nullptr,0,0, nullptr, 0, 1.0f);
  // d_inv = 1 / (qp . ksum)
  dinv_kernel<<<(int)(((ll)NB*NH*LSEQ + 255) / 256), 256, 0, st>>>(w.qp, w.ksum, w.dinv);
  // o = d_inv * (qp @ ctx) -> fused [b, token, h*64+d]
  launch_gemm(st, w.qp, w.ctx, w.attno, LSEQ, DHD, MDIM, MPAD,1, DHD,1, DMOD,1,
              NB, NH, (ll)NH*LSEQ*MPAD, (ll)LSEQ*MPAD, (ll)NH*MPAD*DHD, (ll)MPAD*DHD,
              (ll)LSEQ*DMOD, DHD,
              nullptr, w.dinv, (ll)NH*LSEQ, (ll)LSEQ, nullptr, 0, 1.0f);
  // out projection + residual
  launch_gemm(st, w.attno, outw, io, ROWS, DMOD, DMOD, DMOD,1, DMOD,1, DMOD,1,
              1,1, 0,0,0,0,0,0, outb, nullptr,0,0, io, 0, 1.0f);
}

static void ffn(hipStream_t st, WS& w, const float* xn, float* io,
                const float* w1, const float* b1, const float* w2, const float* b2) {
  launch_gemm(st, xn, w1, w.ffb, ROWS, FFD, DMOD, DMOD,1, FFD,1, FFD,1,
              1,1, 0,0,0,0,0,0, b1, nullptr,0,0, nullptr, 1, 1.0f);
  launch_gemm(st, w.ffb, w2, io, ROWS, DMOD, FFD, FFD,1, DMOD,1, DMOD,1,
              1,1, 0,0,0,0,0,0, b2, nullptr,0,0, io, 0, 1.0f);
}

extern "C" void kernel_launch(void* const* d_in, const int* in_sizes, int n_in,
                              void* d_out, int out_size, void* d_ws, size_t ws_size,
                              hipStream_t stream) {
  (void)in_sizes; (void)n_in; (void)out_size; (void)ws_size;
  const float* x1          = (const float*)d_in[0];
  const float* x2          = (const float*)d_in[1];
  const float* conv_w      = (const float*)d_in[2];
  const float* conv_b      = (const float*)d_in[3];
  const float* embed_w     = (const float*)d_in[4];
  const float* embed_b     = (const float*)d_in[5];
  const float* enc_g       = (const float*)d_in[6];
  const float* enc_qkv_w   = (const float*)d_in[7];
  const float* enc_qkv_b   = (const float*)d_in[8];
  const float* enc_out_w   = (const float*)d_in[9];
  const float* enc_out_b   = (const float*)d_in[10];
  const float* enc_proj    = (const float*)d_in[11];
  const float* enc_ff_w1   = (const float*)d_in[12];
  const float* enc_ff_b1   = (const float*)d_in[13];
  const float* enc_ff_w2   = (const float*)d_in[14];
  const float* enc_ff_b2   = (const float*)d_in[15];
  const float* dec_g       = (const float*)d_in[16];
  const float* dec_s_qkv_w = (const float*)d_in[17];
  const float* dec_s_qkv_b = (const float*)d_in[18];
  const float* dec_s_out_w = (const float*)d_in[19];
  const float* dec_s_out_b = (const float*)d_in[20];
  const float* dec_s_proj  = (const float*)d_in[21];
  const float* dec_c_qkv_w = (const float*)d_in[22];
  const float* dec_c_qkv_b = (const float*)d_in[23];
  const float* dec_c_out_w = (const float*)d_in[24];
  const float* dec_c_out_b = (const float*)d_in[25];
  const float* dec_c_proj  = (const float*)d_in[26];
  const float* dec_ff_w1   = (const float*)d_in[27];
  const float* dec_ff_b1   = (const float*)d_in[28];
  const float* dec_ff_w2   = (const float*)d_in[29];
  const float* dec_ff_b2   = (const float*)d_in[30];
  const float* gen_w       = (const float*)d_in[31];
  const float* gen_b       = (const float*)d_in[32];

  // carve workspace
  char* wp = (char*)d_ws;
  auto carve = [&](ll nfloats) { float* p = (float*)wp; wp += ((nfloats * 4 + 255) / 256) * 256; return p; };
  WS w;
  const ll XSZ = (ll)ROWS * DMOD;
  w.x     = carve(XSZ);
  w.y     = carve(XSZ);
  w.xn    = carve(XSZ);
  w.q     = carve(XSZ);
  w.k     = carve(XSZ);
  w.v     = carve(XSZ);
  w.attno = carve(XSZ);
  w.qp    = carve((ll)NB * NH * LSEQ * MPAD);
  w.kp    = carve((ll)NB * NH * LSEQ * MPAD);
  w.ctx   = carve((ll)NB * NH * MPAD * DHD);
  w.ksum  = carve((ll)NB * NH * MPAD);
  w.dinv  = carve((ll)NB * NH * LSEQ);
  w.ffb   = w.qp;   // reuse: FF buffer (16.8M floats) fits in qp region (17.8M)

  // encoder moderator: conv2d(1->512, k=2) over [L1+1, 2] -> x [B, L1, 512]
  {
    ll total = (ll)NB * LSEQ * DMOD;
    conv_embed_kernel<<<(int)((total + 255) / 256), 256, 0, stream>>>(x1, conv_w, conv_b, w.x);
  }
  // encoder stack
  for (int d = 0; d < NDEPTH; d++) {
    scalenorm_kernel<<<ROWS, 256, 0, stream>>>(w.x, w.xn, enc_g + d * 2 + 0, DMOD);
    attention(stream, w, w.xn, w.xn,
              enc_qkv_w + (ll)d * 3 * DMOD * DMOD, enc_qkv_b + (ll)d * 3 * DMOD,
              enc_out_w + (ll)d * DMOD * DMOD, enc_out_b + (ll)d * DMOD,
              enc_proj + (ll)d * MDIM * DHD, w.x);
    scalenorm_kernel<<<ROWS, 256, 0, stream>>>(w.x, w.xn, enc_g + d * 2 + 1, DMOD);
    ffn(stream, w, w.xn, w.x,
        enc_ff_w1 + (ll)d * DMOD * FFD, enc_ff_b1 + (ll)d * FFD,
        enc_ff_w2 + (ll)d * FFD * DMOD, enc_ff_b2 + (ll)d * DMOD);
  }
  // decoder embedding: y = x2 @ embed_w + embed_b
  launch_gemm(stream, x2, embed_w, w.y, ROWS, DMOD, DSD, DSD,1, DMOD,1, DMOD,1,
              1,1, 0,0,0,0,0,0, embed_b, nullptr,0,0, nullptr, 0, 1.0f);
  // decoder stack
  for (int d = 0; d < NDEPTH; d++) {
    scalenorm_kernel<<<ROWS, 256, 0, stream>>>(w.y, w.xn, dec_g + d * 4 + 0, DMOD);
    attention(stream, w, w.xn, w.xn,
              dec_s_qkv_w + (ll)d * 3 * DMOD * DMOD, dec_s_qkv_b + (ll)d * 3 * DMOD,
              dec_s_out_w + (ll)d * DMOD * DMOD, dec_s_out_b + (ll)d * DMOD,
              dec_s_proj + (ll)d * MDIM * DHD, w.y);
    scalenorm_kernel<<<ROWS, 256, 0, stream>>>(w.y, w.xn, dec_g + d * 4 + 1, DMOD);
    ffn(stream, w, w.xn, w.y,
        dec_ff_w1 + (ll)(d * 2 + 0) * DMOD * FFD, dec_ff_b1 + (ll)(d * 2 + 0) * FFD,
        dec_ff_w2 + (ll)(d * 2 + 0) * FFD * DMOD, dec_ff_b2 + (ll)(d * 2 + 0) * DMOD);
    scalenorm_kernel<<<ROWS, 256, 0, stream>>>(w.y, w.xn, dec_g + d * 4 + 2, DMOD);
    attention(stream, w, w.xn, w.x,   // cross: kv from encoder output (un-normed)
              dec_c_qkv_w + (ll)d * 3 * DMOD * DMOD, dec_c_qkv_b + (ll)d * 3 * DMOD,
              dec_c_out_w + (ll)d * DMOD * DMOD, dec_c_out_b + (ll)d * DMOD,
              dec_c_proj + (ll)d * MDIM * DHD, w.y);
    scalenorm_kernel<<<ROWS, 256, 0, stream>>>(w.y, w.xn, dec_g + d * 4 + 3, DMOD);
    ffn(stream, w, w.xn, w.y,
        dec_ff_w1 + (ll)(d * 2 + 1) * DMOD * FFD, dec_ff_b1 + (ll)(d * 2 + 1) * FFD,
        dec_ff_w2 + (ll)(d * 2 + 1) * FFD * DMOD, dec_ff_b2 + (ll)(d * 2 + 1) * DMOD);
  }
  // generator head: atan(y@gen_w + gen_b)/pi + 0.5 -> d_out [8192, 64]
  launch_gemm(stream, w.y, gen_w, (float*)d_out, ROWS, OSD, DMOD, DMOD,1, OSD,1, OSD,1,
              1,1, 0,0,0,0,0,0, gen_b, nullptr,0,0, nullptr, 3, 1.0f);
}
